// JEPAPointDecoder_43542378447074
// MI455X (gfx1250) — matrix operand order repaired and checked by reading, not compiled, not run on hardware
//
#include <hip/hip_runtime.h>
#include <hip/hip_bf16.h>

// ---------------------------------------------------------------------------
// Types for CDNA5 WMMA (wave32, 16x16x32 f16 -> f32)
// ---------------------------------------------------------------------------
typedef _Float16 half_t;
typedef half_t v16h __attribute__((ext_vector_type(16)));
typedef half_t v8h  __attribute__((ext_vector_type(8)));
typedef float  v8f  __attribute__((ext_vector_type(8)));

__device__ __forceinline__ v8f wmma_f16(v16h a, v16h b, v8f c) {
  // 8 args: (neg_a, A, neg_b, B, c_mod, C, reuse_a, reuse_b)
  return __builtin_amdgcn_wmma_f32_16x16x32_f16(false, a, false, b, (short)0, c,
                                                false, false);
}

// A fragment (16x32 f16, M x K) from LDS. Assumed lane layout (ISA 7.12.2):
// lanes 0-15: M=lane, K 0..7 + 16..23 ; lanes 16-31: M=lane-16, K 8..15 + 24..31.
// ld (in halves) must be a multiple of 8 so each 8-half run is 16B aligned.
__device__ __forceinline__ v16h a_frag_lds(const half_t* s, int ld, int row,
                                           int ks, int lane) {
  const half_t* p = s + (size_t)row * ld + ks * 32 + (lane >> 4) * 8;
  v8h lo = *reinterpret_cast<const v8h*>(p);
  v8h hi = *reinterpret_cast<const v8h*>(p + 16);
  return __builtin_shufflevector(lo, hi, 0, 1, 2, 3, 4, 5, 6, 7,
                                 8, 9, 10, 11, 12, 13, 14, 15);
}

// A fragment from f32 global (rows contiguous, converted on the fly).
__device__ __forceinline__ v16h a_frag_f32(const float* base, int ld, int row,
                                           int ks, int lane) {
  const int k0 = ks * 32 + (lane >> 4) * 8;
  const float* p = base + (size_t)row * ld + k0;
  v16h a;
#pragma unroll
  for (int e = 0; e < 8; ++e) {
    a[e]     = (half_t)p[e];
    a[8 + e] = (half_t)p[16 + e];
  }
  return a;
}

// B fragment from swizzled (fragment-major) weights: per (frag,lane) the 16
// operand halves are contiguous (32B, 32B-aligned) -> 2x global_load_b128.
__device__ __forceinline__ v16h b_frag_swz(const half_t* w, int frag, int lane) {
  return *reinterpret_cast<const v16h*>(w + (((size_t)frag * 32 + lane) << 4));
}

// ---------------------------------------------------------------------------
// Problem constants (from setup_inputs)
// ---------------------------------------------------------------------------
#define BB 4
#define PP 384
#define MM 64
#define CC 128
#define TOK 1024
#define RR 12
#define NCTX (BB * PP * RR)          // 18432
#define NPRED (BB * MM * RR)         // 3072
#define GRP_CTX (PP * RR)            // 4608
#define NALL_G (GRP_CTX + MM * RR)   // 5376 per batch
#define NALL (BB * NALL_G)           // 21504
#define K1PAD 288                    // 262 padded to mult of 32
#define MSG_LD 304                   // halves; 608B rows -> 16B-aligned runs
#define H1_LD 136                    // halves; 272B rows -> 16B-aligned runs

// ---------------------------------------------------------------------------
// Weight convert + swizzle to fragment-major f16:
// dst[((ks*nct + ct)*32 + lane)*16 + e] = src[k, n], n = ct*16 + (lane&15),
// k = ks*32 + (lane>>4)*8 + (e&7) + (e>>3)*16, zero-padded to [Kp, Np].
// ---------------------------------------------------------------------------
__global__ void cvtswz_kernel(const float* __restrict__ src, half_t* __restrict__ dst,
                              int K, int Nn, int Kp, int Np) {
  int t = blockIdx.x * 256 + threadIdx.x;
  if (t >= Kp * Np) return;
  int e    = t & 15;
  int lane = (t >> 4) & 31;
  int frag = t >> 9;
  int nct  = Np >> 4;
  int ks = frag / nct, ct = frag - ks * nct;
  int n = ct * 16 + (lane & 15);
  int k = ks * 32 + (lane >> 4) * 8 + (e & 7) + (e >> 3) * 16;
  float v = (k < K && n < Nn) ? src[(size_t)k * Nn + n] : 0.f;
  dst[t] = (half_t)v;
}

// ---------------------------------------------------------------------------
// Token projection: out[rows,128] = tok[rows,1024] @ Wp + bp   (WMMA)
// 128 threads (4 waves), 64 rows per workgroup.
// ---------------------------------------------------------------------------
__global__ __launch_bounds__(128) void proj_kernel(const float* __restrict__ tok,
                                                   const half_t* __restrict__ Wh,
                                                   const float* __restrict__ bias,
                                                   float* __restrict__ out, int rows) {
  const int lane = threadIdx.x & 31, wid = threadIdx.x >> 5;
  const int rt = blockIdx.x * 64 + wid * 16;
  if (rt >= rows) return;
  const v8f vz = {0.f, 0.f, 0.f, 0.f, 0.f, 0.f, 0.f, 0.f};
  v8f acc[8];
#pragma unroll
  for (int ct = 0; ct < 8; ++ct) acc[ct] = vz;
  const int m = lane & 15;
  for (int ks = 0; ks < TOK / 32; ++ks) {
    v16h a = a_frag_f32(tok, TOK, rt + m, ks, lane);
#pragma unroll
    for (int ct = 0; ct < 8; ++ct) {
      v16h b = b_frag_swz(Wh, ks * 8 + ct, lane);
      acc[ct] = wmma_f16(a, b, acc[ct]);
    }
  }
#pragma unroll
  for (int ct = 0; ct < 8; ++ct) {
    int n = ct * 16 + (lane & 15);
    float bv = bias[n];
#pragma unroll
    for (int r = 0; r < 8; ++r) {
      int mm = r + 8 * (lane >> 4);
      out[(size_t)(rt + mm) * CC + n] = acc[ct][r] + bv;
    }
  }
}

// ---------------------------------------------------------------------------
// anchor[b] = mean_p ctx_xyz[b,p,:]
// ---------------------------------------------------------------------------
__global__ void anchor_kernel(const float* __restrict__ cxyz, float* __restrict__ anchor) {
  __shared__ float sred[3][128];
  int b = blockIdx.x, t = threadIdx.x;
  float ax = 0.f, ay = 0.f, az = 0.f;
  for (int p = t; p < PP; p += 128) {
    const float* q = cxyz + ((size_t)b * PP + p) * 3;
    ax += q[0]; ay += q[1]; az += q[2];
  }
  sred[0][t] = ax; sred[1][t] = ay; sred[2][t] = az;
  __syncthreads();
  for (int s = 64; s > 0; s >>= 1) {
    if (t < s) {
      sred[0][t] += sred[0][t + s];
      sred[1][t] += sred[1][t + s];
      sred[2][t] += sred[2][t + s];
    }
    __syncthreads();
  }
  if (t == 0) {
    anchor[b * 3 + 0] = sred[0][0] * (1.f / PP);
    anchor[b * 3 + 1] = sred[1][0] * (1.f / PP);
    anchor[b * 3 + 2] = sred[2][0] * (1.f / PP);
  }
}

// ---------------------------------------------------------------------------
// xyz = base + unit(noise)*radius
// ---------------------------------------------------------------------------
__global__ void noise_ctx_kernel(const float* __restrict__ cxyz,
                                 const float* __restrict__ noise,
                                 float* __restrict__ out) {
  int t = blockIdx.x * 256 + threadIdx.x;
  if (t >= NCTX) return;
  int b = t / GRP_CTX;
  int rem = t - b * GRP_CTX;
  int p = rem / RR;
  float nx = noise[(size_t)t * 3 + 0], ny = noise[(size_t)t * 3 + 1], nz = noise[(size_t)t * 3 + 2];
  float inv = 0.02f / (sqrtf(nx * nx + ny * ny + nz * nz) + 1e-6f);
  const float* base = cxyz + ((size_t)b * PP + p) * 3;
  out[(size_t)t * 3 + 0] = base[0] + nx * inv;
  out[(size_t)t * 3 + 1] = base[1] + ny * inv;
  out[(size_t)t * 3 + 2] = base[2] + nz * inv;
}

__global__ void noise_pred_kernel(const float* __restrict__ anchor,
                                  const float* __restrict__ noise,
                                  float* __restrict__ out) {
  int t = blockIdx.x * 256 + threadIdx.x;
  if (t >= NPRED) return;
  int b = t / (MM * RR);
  float nx = noise[(size_t)t * 3 + 0], ny = noise[(size_t)t * 3 + 1], nz = noise[(size_t)t * 3 + 2];
  float inv = 0.05f / (sqrtf(nx * nx + ny * ny + nz * nz) + 1e-6f);
  const float* base = anchor + b * 3;
  out[(size_t)t * 3 + 0] = base[0] + nx * inv;
  out[(size_t)t * 3 + 1] = base[1] + ny * inv;
  out[(size_t)t * 3 + 2] = base[2] + nz * inv;
}

// ---------------------------------------------------------------------------
// Broadcast per-token features to R replicas (float4 copies)
// ---------------------------------------------------------------------------
__global__ void bcast_kernel(const float* __restrict__ src, float* __restrict__ dst,
                             int totalExp, int rep) {
  int t = blockIdx.x * 256 + threadIdx.x;   // over totalExp*32 float4
  if (t >= totalExp * 32) return;
  int c4 = t & 31, pr = t >> 5;
  int p = pr / rep;
  reinterpret_cast<float4*>(dst)[(size_t)pr * 32 + c4] =
      reinterpret_cast<const float4*>(src)[(size_t)p * 32 + c4];
}

// ---------------------------------------------------------------------------
// kNN (k=8) within contiguous groups; group xyz staged in LDS.
// grid = nGroups * blocksPerGroup, 256 threads.
// ---------------------------------------------------------------------------
__global__ __launch_bounds__(256) void knn_big_kernel(const float* __restrict__ xyz,
                                                      int* __restrict__ idxOut,
                                                      int groupSize, int blocksPerGroup) {
  __shared__ float s_x[NALL_G * 3];    // 64512 B, covers both 4608 and 5376
  int g = blockIdx.x / blocksPerGroup;
  int chunk = blockIdx.x - g * blocksPerGroup;
  int base = g * groupSize;
  for (int e = threadIdx.x; e < groupSize * 3; e += 256)
    s_x[e] = xyz[(size_t)base * 3 + e];
  __syncthreads();
  int i = chunk * 256 + threadIdx.x;
  if (i >= groupSize) return;
  float xi = s_x[i * 3 + 0], yi = s_x[i * 3 + 1], zi = s_x[i * 3 + 2];
  float bd[8]; int bi[8];
#pragma unroll
  for (int s = 0; s < 8; ++s) { bd[s] = 3.0e38f; bi[s] = 0; }
  for (int j = 0; j < groupSize; ++j) {
    if (j == i) continue;
    float dx = s_x[j * 3 + 0] - xi;
    float dy = s_x[j * 3 + 1] - yi;
    float dz = s_x[j * 3 + 2] - zi;
    float d = dx * dx + dy * dy + dz * dz;
    if (d < bd[7]) {
      bd[7] = d; bi[7] = j;
#pragma unroll
      for (int s = 7; s > 0; --s)
        if (bd[s] < bd[s - 1]) {
          float td = bd[s]; bd[s] = bd[s - 1]; bd[s - 1] = td;
          int ti = bi[s]; bi[s] = bi[s - 1]; bi[s - 1] = ti;
        }
    }
  }
#pragma unroll
  for (int s = 0; s < 8; ++s) idxOut[(size_t)(base + i) * 8 + s] = base + bi[s];
}

// kNN for the 256 tiny 12-point graphs (pred branch), k=min(8,11)=8.
__global__ void knn_small_kernel(const float* __restrict__ xyz, int* __restrict__ idxOut) {
  int t = blockIdx.x * 256 + threadIdx.x;
  if (t >= NPRED) return;
  int g = t / RR, r = t - g * RR, base = g * RR;
  float xi = xyz[(size_t)t * 3], yi = xyz[(size_t)t * 3 + 1], zi = xyz[(size_t)t * 3 + 2];
  float bd[8]; int bi[8];
#pragma unroll
  for (int s = 0; s < 8; ++s) { bd[s] = 3.0e38f; bi[s] = 0; }
  for (int j = 0; j < RR; ++j) {
    if (j == r) continue;
    float dx = xyz[(size_t)(base + j) * 3 + 0] - xi;
    float dy = xyz[(size_t)(base + j) * 3 + 1] - yi;
    float dz = xyz[(size_t)(base + j) * 3 + 2] - zi;
    float d = dx * dx + dy * dy + dz * dz;
    if (d < bd[7]) {
      bd[7] = d; bi[7] = j;
#pragma unroll
      for (int s = 7; s > 0; --s)
        if (bd[s] < bd[s - 1]) {
          float td = bd[s]; bd[s] = bd[s - 1]; bd[s - 1] = td;
          int ti = bi[s]; bi[s] = bi[s - 1]; bi[s - 1] = ti;
        }
    }
  }
#pragma unroll
  for (int s = 0; s < 8; ++s) idxOut[(size_t)t * 8 + s] = base + bi[s];
}

// ---------------------------------------------------------------------------
// EdgeConv: featOut[i] = max_kk( relu([xi, xj-xi] @ W1 + b1) @ W2 + b2 )
// 128 threads / 4 waves, 8 points (64 edge rows) per workgroup. WMMA core.
// ---------------------------------------------------------------------------
__global__ __launch_bounds__(128) void edgeconv_kernel(
    const float* __restrict__ xyz, const float* __restrict__ feat,
    const int* __restrict__ idx, const half_t* __restrict__ W1h,
    const float* __restrict__ b1, const half_t* __restrict__ W2h,
    const float* __restrict__ b2, float* __restrict__ featOut) {
  __shared__ __attribute__((aligned(16))) half_t s_msg[64 * MSG_LD];  // 38912 B
  __shared__ __attribute__((aligned(16))) half_t s_h1[64 * H1_LD];    // 17408 B
  float* s_h2 = reinterpret_cast<float*>(s_msg);  // 64 x 132 f32 (33792 B), reuse

  const int tid = threadIdx.x;
  const int lane = tid & 31, wid = tid >> 5;
  const int p0 = blockIdx.x * 8;

  // stage 0: gather edge messages [xi(131) | xj-xi(131) | pad(26)] as f16
  for (int e = tid; e < 64 * K1PAD; e += 128) {
    int r = e / K1PAD, c = e - r * K1PAD;
    int lp = r >> 3, kk = r & 7;
    int i = p0 + lp;
    int j = idx[(size_t)i * 8 + kk];
    float v;
    if (c < 128)        v = feat[(size_t)i * CC + c];
    else if (c < 131)   v = xyz[(size_t)i * 3 + (c - 128)];
    else if (c < 259) { int cc = c - 131; v = feat[(size_t)j * CC + cc] - feat[(size_t)i * CC + cc]; }
    else if (c < 262) { int cc = c - 259; v = xyz[(size_t)j * 3 + cc] - xyz[(size_t)i * 3 + cc]; }
    else                v = 0.f;
    s_msg[(size_t)r * MSG_LD + c] = (half_t)v;
  }
  __syncthreads();

  const v8f vz = {0.f, 0.f, 0.f, 0.f, 0.f, 0.f, 0.f, 0.f};
  const int rt = wid * 16;
  const int m = lane & 15;

  // stage 1: h1 = relu(msg @ W1 + b1), 9 K-steps x 8 column tiles
  v8f acc[8];
#pragma unroll
  for (int ct = 0; ct < 8; ++ct) acc[ct] = vz;
  for (int ks = 0; ks < K1PAD / 32; ++ks) {
    if (ks < K1PAD / 32 - 1)  // prefetch next k-step's weight fragments
      __builtin_prefetch(W1h + (((size_t)(ks + 1) * 8 * 32 + lane) << 4), 0, 1);
    v16h a = a_frag_lds(s_msg, MSG_LD, rt + m, ks, lane);
#pragma unroll
    for (int ct = 0; ct < 8; ++ct) {
      v16h b = b_frag_swz(W1h, ks * 8 + ct, lane);
      acc[ct] = wmma_f16(a, b, acc[ct]);
    }
  }
#pragma unroll
  for (int ct = 0; ct < 8; ++ct) {
    int n = ct * 16 + (lane & 15);
    float bv = b1[n];
#pragma unroll
    for (int r = 0; r < 8; ++r) {
      int mm = r + 8 * (lane >> 4);
      s_h1[(size_t)(rt + mm) * H1_LD + n] = (half_t)fmaxf(acc[ct][r] + bv, 0.f);
    }
  }
  __syncthreads();   // all waves done reading s_msg, h1 complete

  // stage 2: h2 = h1 @ W2 + b2, 4 K-steps x 8 column tiles
  v8f acc2[8];
#pragma unroll
  for (int ct = 0; ct < 8; ++ct) acc2[ct] = vz;
  for (int ks = 0; ks < CC / 32; ++ks) {
    v16h a = a_frag_lds(s_h1, H1_LD, rt + m, ks, lane);
#pragma unroll
    for (int ct = 0; ct < 8; ++ct) {
      v16h b = b_frag_swz(W2h, ks * 8 + ct, lane);
      acc2[ct] = wmma_f16(a, b, acc2[ct]);
    }
  }
#pragma unroll
  for (int ct = 0; ct < 8; ++ct) {
    int n = ct * 16 + (lane & 15);
    float bv = b2[n];
#pragma unroll
    for (int r = 0; r < 8; ++r) {
      int mm = r + 8 * (lane >> 4);
      s_h2[(size_t)(rt + mm) * 132 + n] = acc2[ct][r] + bv;
    }
  }
  __syncthreads();

  // stage 3: max over the 8 neighbors of each point
  for (int o = tid; o < 8 * CC; o += 128) {
    int lp = o >> 7, c = o & 127;
    float mx = -3.4e38f;
#pragma unroll
    for (int kk = 0; kk < 8; ++kk)
      mx = fmaxf(mx, s_h2[(size_t)(lp * 8 + kk) * 132 + c]);
    featOut[(size_t)(p0 + lp) * CC + c] = mx;
  }
}

// ---------------------------------------------------------------------------
// Offset MLP: xyzOut = xyzIn + relu(feat @ Wa + ba) @ Wb + bb  (Wb padded to 16 cols)
// 128 threads / 4 waves, 64 points per workgroup.
// ---------------------------------------------------------------------------
__global__ __launch_bounds__(128) void offset_kernel(
    const float* __restrict__ feat, const float* __restrict__ xyzIn,
    const half_t* __restrict__ Wah, const float* __restrict__ ba,
    const half_t* __restrict__ Wbh, const float* __restrict__ bb,
    float* __restrict__ xyzOut) {
  __shared__ __attribute__((aligned(16))) half_t s_h1[64 * H1_LD];
  const int lane = threadIdx.x & 31, wid = threadIdx.x >> 5;
  const int p0 = blockIdx.x * 64;
  const int rt = wid * 16;
  const int m = lane & 15;
  const v8f vz = {0.f, 0.f, 0.f, 0.f, 0.f, 0.f, 0.f, 0.f};

  v8f acc[8];
#pragma unroll
  for (int ct = 0; ct < 8; ++ct) acc[ct] = vz;
  for (int ks = 0; ks < CC / 32; ++ks) {
    v16h a = a_frag_f32(feat, CC, p0 + rt + m, ks, lane);
#pragma unroll
    for (int ct = 0; ct < 8; ++ct) {
      v16h b = b_frag_swz(Wah, ks * 8 + ct, lane);
      acc[ct] = wmma_f16(a, b, acc[ct]);
    }
  }
#pragma unroll
  for (int ct = 0; ct < 8; ++ct) {
    int n = ct * 16 + (lane & 15);
    float bv = ba[n];
#pragma unroll
    for (int r = 0; r < 8; ++r) {
      int mm = r + 8 * (lane >> 4);
      s_h1[(size_t)(rt + mm) * H1_LD + n] = (half_t)fmaxf(acc[ct][r] + bv, 0.f);
    }
  }
  __syncthreads();

  v8f acc2 = vz;
  for (int ks = 0; ks < CC / 32; ++ks) {
    v16h a = a_frag_lds(s_h1, H1_LD, rt + m, ks, lane);
    v16h b = b_frag_swz(Wbh, ks, lane);   // nct = 1 (16-wide output)
    acc2 = wmma_f16(a, b, acc2);
  }
  int n = lane & 15;
  if (n < 3) {
    float bv = bb[n];
#pragma unroll
    for (int r = 0; r < 8; ++r) {
      int mm = r + 8 * (lane >> 4);
      int row = p0 + rt + mm;
      xyzOut[(size_t)row * 3 + n] = xyzIn[(size_t)row * 3 + n] + acc2[r] + bv;
    }
  }
}

// ---------------------------------------------------------------------------
// Concatenate ctx + pred into xyz_all / feat_all (per-batch)
// ---------------------------------------------------------------------------
__global__ void concat_kernel(const float* __restrict__ xc, const float* __restrict__ fc,
                              const float* __restrict__ xp, const float* __restrict__ fp,
                              float* __restrict__ xa, float* __restrict__ fa) {
  int t = blockIdx.x * 256 + threadIdx.x;
  if (t < NALL * 32) {   // features, float4 granularity
    int row = t >> 5, c4 = t & 31;
    int b = row / NALL_G, l = row - b * NALL_G;
    const float4* src = (l < GRP_CTX)
        ? reinterpret_cast<const float4*>(fc) + (size_t)(b * GRP_CTX + l) * 32
        : reinterpret_cast<const float4*>(fp) + (size_t)(b * (MM * RR) + (l - GRP_CTX)) * 32;
    reinterpret_cast<float4*>(fa)[(size_t)row * 32 + c4] = src[c4];
  }
  if (t < NALL * 3) {    // xyz
    int row = t / 3, c = t - row * 3;
    int b = row / NALL_G, l = row - b * NALL_G;
    float v = (l < GRP_CTX)
        ? xc[(size_t)(b * GRP_CTX + l) * 3 + c]
        : xp[(size_t)(b * (MM * RR) + (l - GRP_CTX)) * 3 + c];
    xa[(size_t)row * 3 + c] = v;
  }
}

// ---------------------------------------------------------------------------
// Launch
// ---------------------------------------------------------------------------
extern "C" void kernel_launch(void* const* d_in, const int* in_sizes, int n_in,
                              void* d_out, int out_size, void* d_ws, size_t ws_size,
                              hipStream_t stream) {
  (void)in_sizes; (void)n_in; (void)out_size; (void)ws_size;
  const float* ctx_xyz    = (const float*)d_in[0];
  const float* ctx_tokens = (const float*)d_in[1];
  const float* pred_tokens= (const float*)d_in[2];
  const float* noise_ctx  = (const float*)d_in[3];
  const float* noise_pred = (const float*)d_in[4];
  const float* Wp  = (const float*)d_in[5];  const float* bp  = (const float*)d_in[6];
  const float* We1 = (const float*)d_in[7];  const float* be1 = (const float*)d_in[8];
  const float* We2 = (const float*)d_in[9];  const float* be2 = (const float*)d_in[10];
  const float* Wc1 = (const float*)d_in[11]; const float* bc1 = (const float*)d_in[12];
  const float* Wc2 = (const float*)d_in[13]; const float* bc2 = (const float*)d_in[14];
  const float* Wq1 = (const float*)d_in[15]; const float* bq1 = (const float*)d_in[16];
  const float* Wq2 = (const float*)d_in[17]; const float* bq2 = (const float*)d_in[18];
  const float* Wf1 = (const float*)d_in[19]; const float* bf1 = (const float*)d_in[20];
  const float* Wf2 = (const float*)d_in[21]; const float* bf2 = (const float*)d_in[22];

  // workspace carve-out
  size_t off = 0;
  auto alloc = [&](size_t bytes) -> void* {
    void* p = (char*)d_ws + off;
    off += bytes;
    off = (off + 255) & ~(size_t)255;
    return p;
  };
  half_t* Wp16  = (half_t*)alloc((size_t)TOK * CC * 2);
  half_t* W1h   = (half_t*)alloc((size_t)K1PAD * CC * 2);
  half_t* W2h   = (half_t*)alloc((size_t)CC * CC * 2);
  half_t* Wc1h  = (half_t*)alloc((size_t)CC * CC * 2);
  half_t* Wq1h  = (half_t*)alloc((size_t)CC * CC * 2);
  half_t* Wf1h  = (half_t*)alloc((size_t)CC * CC * 2);
  half_t* Wc2h  = (half_t*)alloc((size_t)CC * 16 * 2);
  half_t* Wq2h  = (half_t*)alloc((size_t)CC * 16 * 2);
  half_t* Wf2h  = (half_t*)alloc((size_t)CC * 16 * 2);
  float* featc0   = (float*)alloc((size_t)BB * PP * CC * 4);
  float* featp0   = (float*)alloc((size_t)BB * MM * CC * 4);
  float* anchor   = (float*)alloc((size_t)BB * 3 * 4);
  float* xyzc     = (float*)alloc((size_t)NCTX * 3 * 4);
  float* xyzp     = (float*)alloc((size_t)NPRED * 3 * 4);
  int*   idxc     = (int*)  alloc((size_t)NCTX * 8 * 4);
  int*   idxp     = (int*)  alloc((size_t)NPRED * 8 * 4);
  float* featc_exp= (float*)alloc((size_t)NCTX * CC * 4);
  float* featp_exp= (float*)alloc((size_t)NPRED * CC * 4);
  float* featc1   = (float*)alloc((size_t)NCTX * CC * 4);
  float* featp1   = (float*)alloc((size_t)NPRED * CC * 4);
  float* xyzc1    = (float*)alloc((size_t)NCTX * 3 * 4);
  float* xyzp1    = (float*)alloc((size_t)NPRED * 3 * 4);
  float* xyz_all  = (float*)alloc((size_t)NALL * 3 * 4);
  float* feat_all = (float*)alloc((size_t)NALL * CC * 4);
  int*   idx_all  = (int*)  alloc((size_t)NALL * 8 * 4);
  float* feat_all1= (float*)alloc((size_t)NALL * CC * 4);

  // 1) convert + swizzle weights to fragment-major f16
  cvtswz_kernel<<<(TOK * CC + 255) / 256, 256, 0, stream>>>(Wp,  Wp16, TOK, CC, TOK, CC);
  cvtswz_kernel<<<(K1PAD * CC + 255) / 256, 256, 0, stream>>>(We1, W1h, 262, CC, K1PAD, CC);
  cvtswz_kernel<<<(CC * CC + 255) / 256, 256, 0, stream>>>(We2, W2h, CC, CC, CC, CC);
  cvtswz_kernel<<<(CC * CC + 255) / 256, 256, 0, stream>>>(Wc1, Wc1h, CC, CC, CC, CC);
  cvtswz_kernel<<<(CC * CC + 255) / 256, 256, 0, stream>>>(Wq1, Wq1h, CC, CC, CC, CC);
  cvtswz_kernel<<<(CC * CC + 255) / 256, 256, 0, stream>>>(Wf1, Wf1h, CC, CC, CC, CC);
  cvtswz_kernel<<<(CC * 16 + 255) / 256, 256, 0, stream>>>(Wc2, Wc2h, CC, 3, CC, 16);
  cvtswz_kernel<<<(CC * 16 + 255) / 256, 256, 0, stream>>>(Wq2, Wq2h, CC, 3, CC, 16);
  cvtswz_kernel<<<(CC * 16 + 255) / 256, 256, 0, stream>>>(Wf2, Wf2h, CC, 3, CC, 16);

  // 2) token projections (WMMA GEMM)
  proj_kernel<<<(BB * PP) / 64, 128, 0, stream>>>(ctx_tokens, Wp16, bp, featc0, BB * PP);
  proj_kernel<<<(BB * MM) / 64, 128, 0, stream>>>(pred_tokens, Wp16, bp, featp0, BB * MM);

  // 3) anchor + noisy point clouds
  anchor_kernel<<<BB, 128, 0, stream>>>(ctx_xyz, anchor);
  noise_ctx_kernel<<<(NCTX + 255) / 256, 256, 0, stream>>>(ctx_xyz, noise_ctx, xyzc);
  noise_pred_kernel<<<(NPRED + 255) / 256, 256, 0, stream>>>(anchor, noise_pred, xyzp);

  // 4) broadcast features to replicas
  bcast_kernel<<<(NCTX * 32 + 255) / 256, 256, 0, stream>>>(featc0, featc_exp, NCTX, RR);
  bcast_kernel<<<(NPRED * 32 + 255) / 256, 256, 0, stream>>>(featp0, featp_exp, NPRED, RR);

  // 5) kNN
  knn_big_kernel<<<BB * 18, 256, 0, stream>>>(xyzc, idxc, GRP_CTX, 18);
  knn_small_kernel<<<(NPRED + 255) / 256, 256, 0, stream>>>(xyzp, idxp);

  // 6) EdgeConv (shared We1/We2)
  edgeconv_kernel<<<NCTX / 8, 128, 0, stream>>>(xyzc, featc_exp, idxc, W1h, be1, W2h, be2, featc1);
  edgeconv_kernel<<<NPRED / 8, 128, 0, stream>>>(xyzp, featp_exp, idxp, W1h, be1, W2h, be2, featp1);

  // 7) offset MLPs
  offset_kernel<<<NCTX / 64, 128, 0, stream>>>(featc1, xyzc, Wc1h, bc1, Wc2h, bc2, xyzc1);
  offset_kernel<<<NPRED / 64, 128, 0, stream>>>(featp1, xyzp, Wq1h, bq1, Wq2h, bq2, xyzp1);

  // 8) concat, global kNN + EdgeConv + final offset (writes d_out)
  concat_kernel<<<(NALL * 32 + 255) / 256, 256, 0, stream>>>(xyzc1, featc1, xyzp1, featp1,
                                                             xyz_all, feat_all);
  knn_big_kernel<<<BB * 21, 256, 0, stream>>>(xyz_all, idx_all, NALL_G, 21);
  edgeconv_kernel<<<NALL / 8, 128, 0, stream>>>(xyz_all, feat_all, idx_all, W1h, be1, W2h, be2,
                                                feat_all1);
  offset_kernel<<<NALL / 64, 128, 0, stream>>>(feat_all1, xyz_all, Wf1h, bf1, Wf2h, bf2,
                                               (float*)d_out);
}
